// AgentCrossAttention_34213709480729
// MI455X (gfx1250) — compile-verified, matrix operand order checked by reading, not run on hardware
//
#include <hip/hip_runtime.h>

// ---------------------------------------------------------------------------
// Types for CDNA5 WMMA
// ---------------------------------------------------------------------------
typedef __bf16 bf16_t;
typedef __attribute__((ext_vector_type(16))) __bf16 v16bf;
typedef __attribute__((ext_vector_type(2)))  __bf16 bf16x2;
typedef __attribute__((ext_vector_type(8)))  float  v8f;

// pack two f32 -> two bf16 in one dword using HW conversion
static __device__ __forceinline__ unsigned pack2bf(float f0, float f1) {
#if __has_builtin(__builtin_amdgcn_cvt_pk_bf16_f32)
    bf16x2 t = __builtin_amdgcn_cvt_pk_bf16_f32(f0, f1);
    return __builtin_bit_cast(unsigned, t);
#else
    bf16x2 t;
    t.x = (__bf16)f0;
    t.y = (__bf16)f1;
    return __builtin_bit_cast(unsigned, t);
#endif
}

// clamped + selected load (no exec-mask branches)
static __device__ __forceinline__ float ld_guard(const float* __restrict__ P, long rs, long cs,
                                                 int r, int c, int R, int C) {
    const int rc = r < R ? r : R - 1;
    const int cc = c < C ? c : C - 1;
    const float v = P[(long)rc * rs + (long)cc * cs];
    return (r < R && c < C) ? v : 0.0f;
}

#define BM 64
#define BN 64
#define BK 32
#define LDSW 20   // dwords per LDS row (16 data + 4 pad); 80B rows, 16B-aligned

// stage one 64-row x 32-k tile (f32 -> packed bf16 dwords in LDS)
template <bool GUARD, bool BVEC>
static __device__ __forceinline__ void stage_tiles(
    const float* __restrict__ A, int lda,
    const float* __restrict__ B, int B_rs, int B_cs,
    int M, int N, int K, int m0, int n0, int k0,
    int lrow, int lk, int j0,
    unsigned* __restrict__ As, unsigned* __restrict__ Bs)
{
    // ---- A tile ----
    {
        const int gm = m0 + lrow;
        const int gk = k0 + lk;
        uint4 pk;
        if (!GUARD || (gm < M && gk + 7 < K)) {
            const float* ap = A + (long)gm * lda + gk;
            const float4 a0 = *(const float4*)ap;
            const float4 a1 = *(const float4*)(ap + 4);
            pk.x = pack2bf(a0.x, a0.y);
            pk.y = pack2bf(a0.z, a0.w);
            pk.z = pack2bf(a1.x, a1.y);
            pk.w = pack2bf(a1.z, a1.w);
            if (k0 + BK < K) __builtin_prefetch(ap + BK, 0, 1);
        } else {
            pk.x = pack2bf(ld_guard(A, lda, 1, gm, gk + 0, M, K),
                           ld_guard(A, lda, 1, gm, gk + 1, M, K));
            pk.y = pack2bf(ld_guard(A, lda, 1, gm, gk + 2, M, K),
                           ld_guard(A, lda, 1, gm, gk + 3, M, K));
            pk.z = pack2bf(ld_guard(A, lda, 1, gm, gk + 4, M, K),
                           ld_guard(A, lda, 1, gm, gk + 5, M, K));
            pk.w = pack2bf(ld_guard(A, lda, 1, gm, gk + 6, M, K),
                           ld_guard(A, lda, 1, gm, gk + 7, M, K));
        }
        *(uint4*)&As[lrow * LDSW + j0] = pk;
    }
    // ---- B tile ----
    {
        const int gn = n0 + lrow;
        const int gk = k0 + lk;
        uint4 pk;
        if (BVEC && (!GUARD || (gn < N && gk + 7 < K))) {
            const float* bp = B + (long)gn * B_rs + gk;
            const float4 b0 = *(const float4*)bp;
            const float4 b1 = *(const float4*)(bp + 4);
            pk.x = pack2bf(b0.x, b0.y);
            pk.y = pack2bf(b0.z, b0.w);
            pk.z = pack2bf(b1.x, b1.y);
            pk.w = pack2bf(b1.z, b1.w);
            if (k0 + BK < K) __builtin_prefetch(bp + BK, 0, 1);
        } else {
            pk.x = pack2bf(ld_guard(B, B_rs, B_cs, gn, gk + 0, N, K),
                           ld_guard(B, B_rs, B_cs, gn, gk + 1, N, K));
            pk.y = pack2bf(ld_guard(B, B_rs, B_cs, gn, gk + 2, N, K),
                           ld_guard(B, B_rs, B_cs, gn, gk + 3, N, K));
            pk.z = pack2bf(ld_guard(B, B_rs, B_cs, gn, gk + 4, N, K),
                           ld_guard(B, B_rs, B_cs, gn, gk + 5, N, K));
            pk.w = pack2bf(ld_guard(B, B_rs, B_cs, gn, gk + 6, N, K),
                           ld_guard(B, B_rs, B_cs, gn, gk + 7, N, K));
        }
        *(uint4*)&Bs[lrow * LDSW + j0] = pk;
    }
}

// ---------------------------------------------------------------------------
// Generic batched GEMM:  C[m,n] = alpha * sum_k A[m,k]*B[n,k]  (+ bias[m,n])
//   A: row-major (k-stride 1), B: strides (B_rs, B_cs)
//   GUARD: bounds checks needed. BVEC: B_cs==1 (vectorizable).
// Block: 256 threads = 8 waves (4 M-strips x 2 N-halves). Tile 64x64, K-step 32.
// Double-buffered LDS: one barrier per K-iteration.
// ---------------------------------------------------------------------------
template <bool GUARD, bool BVEC>
__global__ __launch_bounds__(256)
void gemm_bf16_wmma(const float* __restrict__ A, int lda, long A_z0, long A_z1,
                    const float* __restrict__ B, int B_rs, int B_cs, long B_z0, long B_z1,
                    float* __restrict__ C, int ldc, long C_z0, long C_z1,
                    const float* __restrict__ bias, int bias_rs, long bias_z0, long bias_z1,
                    int M, int N, int K, float alpha, int nZ1)
{
    __shared__ unsigned As[2][BM * LDSW];
    __shared__ unsigned Bs[2][BN * LDSW];

    const int z  = blockIdx.z;
    const int z0 = z / nZ1;
    const int z1 = z % nZ1;
    A += (long)z0 * A_z0 + (long)z1 * A_z1;
    B += (long)z0 * B_z0 + (long)z1 * B_z1;
    C += (long)z0 * C_z0 + (long)z1 * C_z1;
    if (bias) bias += (long)z0 * bias_z0 + (long)z1 * bias_z1;

    const int m0 = blockIdx.y * BM;
    const int n0 = blockIdx.x * BN;

    const int tid  = threadIdx.x;
    const int lane = tid & 31;
    const int wave = tid >> 5;
    const int wm   = wave & 3;    // M strip (16 rows)
    const int wn   = wave >> 2;   // N half (32 cols)
    const int half = lane >> 4;
    const int l16  = lane & 15;

    v8f acc0 = {};
    v8f acc1 = {};

    // loader: each thread fills 8 contiguous k (= 4 packed dwords) of one row
    const int lrow = tid >> 2;         // 0..63
    const int lk   = (tid & 3) * 8;    // k offset 0,8,16,24
    const int j0   = (tid & 3) * 4;    // dword offset 0,4,8,12

    const int nk = (K + BK - 1) / BK;

    stage_tiles<GUARD, BVEC>(A, lda, B, B_rs, B_cs, M, N, K, m0, n0, 0,
                             lrow, lk, j0, As[0], Bs[0]);
    __syncthreads();

    const int am  = wm * 16 + l16;
    const int bn0 = wn * 32 + l16;
    const int bn1 = bn0 + 16;

    for (int t = 0; t < nk; ++t) {
        const int cur = t & 1;
        if (t + 1 < nk) {
            stage_tiles<GUARD, BVEC>(A, lda, B, B_rs, B_cs, M, N, K, m0, n0, (t + 1) * BK,
                                     lrow, lk, j0, As[cur ^ 1], Bs[cur ^ 1]);
        }
        // fragments per ISA 16-bit 16x32 layout:
        // lane (row = L%16, half = L/16); vgpr i holds pair at dword
        //   jd = (i%4) + half*4 + (i/4)*8
        union FragU { v16bf v; unsigned u[8]; };
        FragU af, bf0, bf1;
#pragma unroll
        for (int i = 0; i < 8; ++i) {
            const int jd = (i & 3) + half * 4 + (i >> 2) * 8;
            af.u[i]  = As[cur][am  * LDSW + jd];
            bf0.u[i] = Bs[cur][bn0 * LDSW + jd];
            bf1.u[i] = Bs[cur][bn1 * LDSW + jd];
        }
        acc0 = __builtin_amdgcn_wmma_f32_16x16x32_bf16(false, af.v, false, bf0.v,
                                                       (short)0, acc0, false, false);
        acc1 = __builtin_amdgcn_wmma_f32_16x16x32_bf16(false, af.v, false, bf1.v,
                                                       (short)0, acc1, false, false);
        __syncthreads();
    }

    // epilogue: C layout -> lane n = L%16 ; vgpr r -> m = r + half*8
#pragma unroll
    for (int r = 0; r < 8; ++r) {
        const int gm = m0 + wm * 16 + r + half * 8;
        if (GUARD && gm >= M) continue;
        const int gn0 = n0 + wn * 32 + l16;
        const int gn1 = gn0 + 16;
        if (!GUARD || gn0 < N) {
            float v = alpha * acc0[r];
            if (bias) v += bias[(long)gm * bias_rs + gn0];
            C[(long)gm * ldc + gn0] = v;
        }
        if (!GUARD || gn1 < N) {
            float v = alpha * acc1[r];
            if (bias) v += bias[(long)gm * bias_rs + gn1];
            C[(long)gm * ldc + gn1] = v;
        }
    }
}

// ---------------------------------------------------------------------------
// Pooling: q[b,4096,512] (+ context[b,4096,256]) -> pooled [b][768][49]
// ---------------------------------------------------------------------------
__global__ __launch_bounds__(256)
void pool_kernel(const float* __restrict__ q, const float* __restrict__ ctx,
                 float* __restrict__ pooled)
{
    const long total = 16L * 768 * 49;
    long idx = (long)blockIdx.x * 256 + threadIdx.x;
    if (idx >= total) return;
    const int pq = (int)(idx % 49);
    const int c  = (int)((idx / 49) % 768);
    const int b  = (int)(idx / (49L * 768));
    const int pi = pq / 7, pj = pq % 7;
    const int hs = pi * 64 / 7, he = ((pi + 1) * 64 + 6) / 7;
    const int ws = pj * 64 / 7, we = ((pj + 1) * 64 + 6) / 7;
    const float* src;
    int stride;
    if (c < 512) { src = q   + (long)b * 4096 * 512 + c;         stride = 512; }
    else         { src = ctx + (long)b * 4096 * 256 + (c - 512); stride = 256; }
    float sum = 0.0f;
    for (int h = hs; h < he; ++h)
        for (int w = ws; w < we; ++w)
            sum += src[(long)(h * 64 + w) * stride];
    pooled[idx] = sum / (float)((he - hs) * (we - ws));
}

// ---------------------------------------------------------------------------
// agent = conv2d(pooled[b,768,7,7], w[512,768,3,3], SAME) -> scatter to
// agentT[b][h][a][d]
// ---------------------------------------------------------------------------
__global__ __launch_bounds__(256)
void conv_agent_kernel(const float* __restrict__ pooled, const float* __restrict__ w,
                       const float* __restrict__ bias, float* __restrict__ agentT)
{
    const long total = 16L * 512 * 49;
    long idx = (long)blockIdx.x * 256 + threadIdx.x;
    if (idx >= total) return;
    const int pos = (int)(idx % 49);
    const int oc  = (int)((idx / 49) % 512);
    const int b   = (int)(idx / (49L * 512));
    const int py = pos / 7, px = pos % 7;
    float sum = bias[oc];
    const float* wp = w + (long)oc * 768 * 9;
    const float* ip = pooled + (long)b * 768 * 49;
    for (int ic = 0; ic < 768; ++ic) {
        const float* wpc = wp + ic * 9;
        const float* ipc = ip + ic * 49;
#pragma unroll
        for (int ky = 0; ky < 3; ++ky) {
            const int y = py + ky - 1;
            if (y < 0 || y > 6) continue;
#pragma unroll
            for (int kx = 0; kx < 3; ++kx) {
                const int x = px + kx - 1;
                if (x < 0 || x > 6) continue;
                sum += wpc[ky * 3 + kx] * ipc[y * 7 + x];
            }
        }
    }
    const int flat = oc * 49 + pos;
    const int a = flat / 512;
    const int rem = flat % 512;
    const int h = rem / 64, d = rem % 64;
    agentT[(((long)b * 8 + h) * 49 + a) * 64 + d] = sum;
}

// ---------------------------------------------------------------------------
// Bilinear 7x7 -> 64x64 (half-pixel centers) + axis bias tables
// ---------------------------------------------------------------------------
static __device__ __forceinline__ float bilin7(const float* __restrict__ src, int oy, int ox)
{
    const float sy = (oy + 0.5f) * (7.0f / 64.0f) - 0.5f;
    const float sx = (ox + 0.5f) * (7.0f / 64.0f) - 0.5f;
    int y0 = (int)floorf(sy); const float fy = sy - (float)y0;
    int x0 = (int)floorf(sx); const float fx = sx - (float)x0;
    int y1 = y0 + 1, x1 = x0 + 1;
    y0 = y0 < 0 ? 0 : (y0 > 6 ? 6 : y0);
    y1 = y1 < 0 ? 0 : (y1 > 6 ? 6 : y1);
    x0 = x0 < 0 ? 0 : (x0 > 6 ? 6 : x0);
    x1 = x1 < 0 ? 0 : (x1 > 6 ? 6 : x1);
    const float t   = src[y0 * 7 + x0] * (1.0f - fx) + src[y0 * 7 + x1] * fx;
    const float btm = src[y1 * 7 + x0] * (1.0f - fx) + src[y1 * 7 + x1] * fx;
    return t * (1.0f - fy) + btm * fy;
}

__global__ __launch_bounds__(256)
void biasA_kernel(const float* __restrict__ an, const float* __restrict__ ah,
                  const float* __restrict__ aw, float* __restrict__ pbA)
{
    const long total = 8L * 49 * 4096;
    long idx = (long)blockIdx.x * 256 + threadIdx.x;
    if (idx >= total) return;
    const int n = (int)(idx % 4096);
    const int a = (int)((idx / 4096) % 49);
    const int h = (int)(idx / (4096L * 49));
    const int hh = n / 64, ww = n % 64;
    float v = bilin7(an + ((long)h * 49 + a) * 49, hh, ww);
    v += ah[((long)h * 49 + a) * 64 + hh];   // ah_bias (1,8,49,64,1)
    v += aw[((long)h * 49 + a) * 64 + ww];   // aw_bias (1,8,49,1,64)
    pbA[idx] = v;
}

__global__ __launch_bounds__(256)
void biasQ_kernel(const float* __restrict__ na, const float* __restrict__ ha,
                  const float* __restrict__ wa, float* __restrict__ abQ)
{
    const long total = 8L * 4096 * 49;
    long idx = (long)blockIdx.x * 256 + threadIdx.x;
    if (idx >= total) return;
    const int a = (int)(idx % 49);
    const int n = (int)((idx / 49) % 4096);
    const int h = (int)(idx / (49L * 4096));
    const int hh = n / 64, ww = n % 64;
    float v = bilin7(na + ((long)h * 49 + a) * 49, hh, ww);
    v += ha[((long)h * 64 + hh) * 49 + a];   // ha_bias (1,8,64,1,49)
    v += wa[((long)h * 64 + ww) * 49 + a];   // wa_bias (1,8,1,64,49)
    abQ[idx] = v;
}

// ---------------------------------------------------------------------------
// Softmax over long rows (ncols=4096), one block per row, in place.
// ---------------------------------------------------------------------------
__global__ __launch_bounds__(256)
void softmax_rows_kernel(float* __restrict__ S, int ncols)
{
    __shared__ float red[8];
    __shared__ float bval;
    float* p = S + (long)blockIdx.x * ncols;
    const int tid = threadIdx.x;

    float m = -3.0e38f;
    for (int i = tid; i < ncols; i += 256) m = fmaxf(m, p[i]);
#pragma unroll
    for (int o = 16; o > 0; o >>= 1) m = fmaxf(m, __shfl_xor(m, o, 32));
    if ((tid & 31) == 0) red[tid >> 5] = m;
    __syncthreads();
    if (tid == 0) { float t = red[0]; for (int i = 1; i < 8; ++i) t = fmaxf(t, red[i]); bval = t; }
    __syncthreads();
    m = bval;

    float s = 0.0f;
    for (int i = tid; i < ncols; i += 256) { float e = __expf(p[i] - m); p[i] = e; s += e; }
#pragma unroll
    for (int o = 16; o > 0; o >>= 1) s += __shfl_xor(s, o, 32);
    if ((tid & 31) == 0) red[tid >> 5] = s;
    __syncthreads();
    if (tid == 0) { float t = 0.0f; for (int i = 0; i < 8; ++i) t += red[i]; bval = t; }
    __syncthreads();
    const float inv = 1.0f / bval;
    for (int i = tid; i < ncols; i += 256) p[i] *= inv;
}

// ---------------------------------------------------------------------------
// Softmax over 49-wide rows stored with leading dim 64; one wave per row.
// ---------------------------------------------------------------------------
__global__ __launch_bounds__(256)
void softmax_rows49_kernel(float* __restrict__ S, long nrows)
{
    const long row = (long)blockIdx.x * 8 + (threadIdx.x >> 5);
    if (row >= nrows) return;
    const int lane = threadIdx.x & 31;
    float* p = S + row * 64;
    const float v0 = p[lane];
    const float v1 = (lane + 32 < 49) ? p[lane + 32] : -3.0e38f;
    float m = fmaxf(v0, v1);
#pragma unroll
    for (int o = 16; o > 0; o >>= 1) m = fmaxf(m, __shfl_xor(m, o, 32));
    const float e0 = __expf(v0 - m);
    const float e1 = (lane + 32 < 49) ? __expf(v1 - m) : 0.0f;
    float s = e0 + e1;
#pragma unroll
    for (int o = 16; o > 0; o >>= 1) s += __shfl_xor(s, o, 32);
    const float inv = 1.0f / s;
    p[lane] = e0 * inv;
    if (lane + 32 < 49) p[lane + 32] = e1 * inv;
}

// ---------------------------------------------------------------------------
// Depthwise 3x3 conv over v (from kv[...,512:]) accumulated into out_pre.
// ---------------------------------------------------------------------------
__global__ __launch_bounds__(256)
void dwc_kernel(const float* __restrict__ kv, const float* __restrict__ w,
                const float* __restrict__ bias, float* __restrict__ out_pre)
{
    const long total = 16L * 4096 * 512;
    long idx = (long)blockIdx.x * 256 + threadIdx.x;
    if (idx >= total) return;
    const int c = (int)(idx % 512);
    const int n = (int)((idx / 512) % 4096);
    const int b = (int)(idx / (512L * 4096));
    const int hy = n / 64, wx = n % 64;
    const float* wp = w + c * 9;
    float sum = bias[c];
#pragma unroll
    for (int ky = 0; ky < 3; ++ky) {
        const int y = hy + ky - 1;
        if (y < 0 || y > 63) continue;
#pragma unroll
        for (int kx = 0; kx < 3; ++kx) {
            const int x = wx + kx - 1;
            if (x < 0 || x > 63) continue;
            sum += wp[ky * 3 + kx] * kv[((long)b * 4096 + y * 64 + x) * 1024 + 512 + c];
        }
    }
    out_pre[idx] += sum;
}

// ---------------------------------------------------------------------------
// Host-side orchestration
// ---------------------------------------------------------------------------
extern "C" void kernel_launch(void* const* d_in, const int* in_sizes, int n_in,
                              void* d_out, int out_size, void* d_ws, size_t ws_size,
                              hipStream_t stream)
{
    (void)in_sizes; (void)n_in; (void)out_size; (void)ws_size;

    const float* x       = (const float*)d_in[0];
    const float* context = (const float*)d_in[1];
    const float* q_w     = (const float*)d_in[2];
    const float* kv_w    = (const float*)d_in[3];
    const float* proj_w  = (const float*)d_in[4];
    const float* proj_b  = (const float*)d_in[5];
    const float* conv_w  = (const float*)d_in[6];
    const float* conv_b  = (const float*)d_in[7];
    const float* dwc_w   = (const float*)d_in[8];
    const float* dwc_b   = (const float*)d_in[9];
    const float* an_bias = (const float*)d_in[10];
    const float* na_bias = (const float*)d_in[11];
    const float* ah_bias = (const float*)d_in[12];
    const float* aw_bias = (const float*)d_in[13];
    const float* ha_bias = (const float*)d_in[14];
    const float* wa_bias = (const float*)d_in[15];
    float* out = (float*)d_out;
    float* ws  = (float*)d_ws;

    const int B = 16, N = 4096, INNER = 512, AG = 49, HD = 64, H = 8;
    const long BN_ = (long)B * N;           // 65536 rows
    const float scale = 0.125f;             // 64^-0.5

    long off = 0;
    float* q       = ws + off; off += BN_ * INNER;
    float* kv      = ws + off; off += BN_ * 2 * INNER;
    float* pooled  = ws + off; off += (long)B * 768 * 49;
    float* agentT  = ws + off; off += (long)B * H * AG * HD;
    float* pbA     = ws + off; off += (long)H * AG * N;
    float* abQ     = ws + off; off += (long)H * N * AG;
    float* agent_v = ws + off; off += (long)B * H * AG * HD;
    float* S       = ws + off; off += (long)B * H * N * 64;
    float* out_pre = ws + off; off += BN_ * INNER;

    // 1) q = x @ q_w^T
    gemm_bf16_wmma<false, true><<<dim3(INNER / 64, BN_ / 64, 1), 256, 0, stream>>>(
        x, 512, 0, 0,  q_w, 512, 1, 0, 0,  q, 512, 0, 0,
        nullptr, 0, 0, 0,  (int)BN_, INNER, 512, 1.0f, 1);

    // 2) kv = x @ kv_w^T
    gemm_bf16_wmma<false, true><<<dim3(1024 / 64, BN_ / 64, 1), 256, 0, stream>>>(
        x, 512, 0, 0,  kv_w, 512, 1, 0, 0,  kv, 1024, 0, 0,
        nullptr, 0, 0, 0,  (int)BN_, 1024, 512, 1.0f, 1);

    // 3) pooling
    {
        long total = 16L * 768 * 49;
        pool_kernel<<<dim3((unsigned)((total + 255) / 256)), 256, 0, stream>>>(q, context, pooled);
    }

    // 4) conv 3x3 -> agentT
    {
        long total = 16L * 512 * 49;
        conv_agent_kernel<<<dim3((unsigned)((total + 255) / 256)), 256, 0, stream>>>(
            pooled, conv_w, conv_b, agentT);
    }

    // 5) position biases
    {
        long tA = 8L * 49 * 4096;
        biasA_kernel<<<dim3((unsigned)((tA + 255) / 256)), 256, 0, stream>>>(an_bias, ah_bias, aw_bias, pbA);
        long tQ = 8L * 4096 * 49;
        biasQ_kernel<<<dim3((unsigned)((tQ + 255) / 256)), 256, 0, stream>>>(na_bias, ha_bias, wa_bias, abQ);
    }

    // 6) scores1: S[bh,49,4096] = scale*agentT @ Kh^T + pbA[h]
    gemm_bf16_wmma<true, true><<<dim3(N / 64, 1, B * H), 256, 0, stream>>>(
        agentT, 64, (long)H * AG * HD, (long)AG * HD,
        kv, 1024, 1, (long)N * 1024, 64,
        S, N, (long)H * AG * N, (long)AG * N,
        pbA, N, 0, (long)AG * N,
        AG, N, HD, scale, H);

    // 7) softmax over n
    softmax_rows_kernel<<<dim3((unsigned)(B * H * AG)), 256, 0, stream>>>(S, N);

    // 8) agent_v = P1 @ Vh
    gemm_bf16_wmma<true, false><<<dim3(1, 1, B * H), 256, 0, stream>>>(
        S, N, (long)H * AG * N, (long)AG * N,
        kv + 512, 1, 1024, (long)N * 1024, 64,
        agent_v, HD, (long)H * AG * HD, (long)AG * HD,
        nullptr, 0, 0, 0,
        AG, HD, N, 1.0f, H);

    // 9) scores2: S[bh,4096,64] = scale*qh @ agentT^T + abQ[h]
    gemm_bf16_wmma<true, true><<<dim3(1, N / 64, B * H), 256, 0, stream>>>(
        q, 512, (long)N * 512, 64,
        agentT, 64, 1, (long)H * AG * HD, (long)AG * HD,
        S, 64, (long)H * N * 64, (long)N * 64,
        abQ, AG, 0, (long)N * AG,
        N, AG, HD, scale, H);

    // 10) softmax over a
    softmax_rows49_kernel<<<dim3((unsigned)((B * H * N) / 8)), 256, 0, stream>>>(S, (long)B * H * N);

    // 11) out_pre = P2 @ agent_v
    gemm_bf16_wmma<true, false><<<dim3(1, N / 64, B * H), 256, 0, stream>>>(
        S, 64, (long)H * N * 64, (long)N * 64,
        agent_v, 1, 64, (long)H * AG * HD, (long)AG * HD,
        out_pre, 512, (long)N * 512, 64,
        nullptr, 0, 0, 0,
        N, HD, AG, 1.0f, H);

    // 12) depthwise conv added into out_pre
    {
        long total = 16L * 4096 * 512;
        dwc_kernel<<<dim3((unsigned)((total + 255) / 256)), 256, 0, stream>>>(kv, dwc_w, dwc_b, out_pre);
    }

    // 13) out = out_pre @ proj_w^T + proj_b
    gemm_bf16_wmma<false, true><<<dim3(INNER / 64, BN_ / 64, 1), 256, 0, stream>>>(
        out_pre, 512, 0, 0,  proj_w, 512, 1, 0, 0,  out, 512, 0, 0,
        proj_b, 0, 0, 0,  (int)BN_, INNER, 512, 1.0f, 1);
}
